// LSTM_21672404976016
// MI455X (gfx1250) — matrix operand order repaired and checked by reading, not compiled, not run on hardware
//
#include <hip/hip_runtime.h>
#include <hip/hip_bf16.h>
#include <math.h>

// Problem sizes (match reference)
#define B_    64
#define T_    1024
#define K_    512           // INPUT_SIZE == HIDDEN_SIZE
#define NG    4
#define NCOL  (NG * K_)     // 2048 gate-concat columns
#define BT_   (B_ * T_)     // 65536 rows of x

// Recurrent phase config
#define NWG_REC 16
#define HSLICE  (K_ / NWG_REC)   // 32 hidden columns per WG

// LDS layout for recurrent kernel
#define LDS_W_BYTES   (128 * 1024)           // W_h slice, fragment-major
#define LDS_XBUF_OFF  LDS_W_BYTES
#define LDS_XBUF_BYTES 32768                 // one step's xproj tile (64x128 f32)
#define LDS_TOTAL     (LDS_W_BYTES + 2 * LDS_XBUF_BYTES)   // 192 KB < 320 KB/WGP

typedef __bf16 bf16;
typedef __attribute__((ext_vector_type(16))) __bf16        v16bf;
typedef __attribute__((ext_vector_type(8)))  float         v8f;
typedef __attribute__((ext_vector_type(4)))  unsigned int  u32x4;
typedef __attribute__((ext_vector_type(4)))  unsigned int  uint32x4;
typedef __attribute__((ext_vector_type(8)))  int           int32x8;
typedef __attribute__((ext_vector_type(4)))  int           int32x4;

union FragU { v16bf v; u32x4 u[2]; };

// ---- WMMA fragment helpers (CDNA5 wave32 layouts, CDNA5 ISA) ----

// 16-bit A matrix 16x32 (MxK), row-major source [m][k]:
// lane<16: m=lane, K={k0..k0+7, k0+16..k0+23}; lane>=16: K shifted by 8.
__device__ __forceinline__ v16bf load_a_frag(const bf16* base, int ldk, int m0,
                                             int k0, int lane) {
  int m = m0 + (lane & 15);
  int k = k0 + ((lane >> 4) << 3);
  const bf16* p = base + (size_t)m * ldk + k;
  FragU f;
  f.u[0] = *(const u32x4*)(p);        // k   .. k+7
  f.u[1] = *(const u32x4*)(p + 16);   // k+16.. k+23
  return f.v;
}

// Fragment-major B source: [frag = (col/16)*16 + k/32][lane][16 bf16].
// Lane-consecutive 32B chunks -> fully coalesced global loads.
__device__ __forceinline__ v16bf load_b_frag_fm(const bf16* fm, int c0, int k0,
                                                int lane) {
  size_t frag = (size_t)(c0 >> 4) * 16 + (k0 >> 5);
  const u32x4* p = (const u32x4*)(fm + (frag * 32 + lane) * 16);
  FragU f;
  f.u[0] = p[0];
  f.u[1] = p[1];
  return f.v;
}

__device__ __forceinline__ v8f wmma_bf16(v16bf a, v16bf b, v8f c) {
  return __builtin_amdgcn_wmma_f32_16x16x32_bf16(false, a, false, b,
                                                 (short)0, c, false, false);
}

__device__ __forceinline__ float sigmoidf_(float x) {
  return 1.0f / (1.0f + __expf(-x));
}

// ---- Tensor Data Mover: build D# per CDNA5 ISA ------
// Up-to-3D tile load Global->LDS. This toolchain uses the 6-arg builtin:
// (uint32x4 g0, int32x8 g1, int32x4 g2, int32x4 g3, int32x8 extra, i32 cpol).
// addr(x,y,z) = global_addr + data_size*(x + y*stride0 + z*stride1)
__device__ __forceinline__ void tdm_load(unsigned lds_off, const void* gp,
                                         int ds_code,           // 0:1B 1:2B 2:4B 3:8B
                                         unsigned td0, unsigned td1, unsigned td2,
                                         unsigned tile0, unsigned tile1,
                                         unsigned tile2,
                                         unsigned long long s0,
                                         unsigned long long s1) {
  unsigned long long ga = (unsigned long long)gp;
  uint32x4 g0;
  g0[0] = 1u;                                   // count=1, user descriptor
  g0[1] = lds_off;                              // LDS byte address
  g0[2] = (unsigned)ga;                         // global_addr[31:0]
  g0[3] = (unsigned)((ga >> 32) & 0x01FFFFFFu) | (2u << 30);  // ga[56:32], type=2
  int32x8 g1;
  g1[0] = (int)((unsigned)ds_code << 16);       // wg_mask=0, data_size
  g1[1] = (int)((td0 & 0xFFFFu) << 16);         // tensor_dim0[15:0]
  g1[2] = (int)(((td1 & 0xFFFFu) << 16) | ((td0 >> 16) & 0xFFFFu));
  g1[3] = (int)(((tile0 & 0xFFFFu) << 16) | ((td1 >> 16) & 0xFFFFu));
  g1[4] = (int)(((tile2 & 0xFFFFu) << 16) | (tile1 & 0xFFFFu));
  g1[5] = (int)(s0 & 0xFFFFFFFFu);              // tensor_dim0_stride[31:0]
  g1[6] = (int)(((s1 & 0xFFFFu) << 16) | ((s0 >> 32) & 0xFFFFu));
  g1[7] = (int)((s1 >> 16) & 0xFFFFFFFFu);
  int32x4 g2;
  g2[0] = (int)td2;                             // tensor_dim2 (z bound)
  g2[1] = 0; g2[2] = 0; g2[3] = 0;
  int32x4 g3 = {0, 0, 0, 0};
  int32x8 gx = {0, 0, 0, 0, 0, 0, 0, 0};
  __builtin_amdgcn_tensor_load_to_lds(g0, g1, g2, g3, gx, 0);
}

// ---- small setup kernels -------------------------------------------------

struct WPtrs { const float* w[8]; };           // 0..3 = W_i gates, 4..7 = W_h gates
struct BPtrs { const float* bi[4]; const float* bh[4]; };

__global__ void init_state(unsigned int* hzero, unsigned int* syncc) {
  int idx = blockIdx.x * blockDim.x + threadIdx.x;
  if (idx < (2 * B_ * K_ * 2) / 4) hzero[idx] = 0u;   // both h ping-pong buffers
  if (idx == 0) *syncc = 0u;
}

__global__ void cvt_x(const float* __restrict__ x, bf16* __restrict__ xb) {
  size_t i = ((size_t)blockIdx.x * blockDim.x + threadIdx.x) * 4;
  const float4 v = *(const float4*)(x + i);
  bf16* o = xb + i;
  o[0] = (bf16)v.x; o[1] = (bf16)v.y; o[2] = (bf16)v.z; o[3] = (bf16)v.w;
}

// [k][n] f32 -> fragment-major bf16: [(gate*32 + n/16)*16 + k/32][lane][16]
// lane = (n%16) | ((k/8)%2)<<4 ; element j = (k%8) | ((k%32)/16)<<3.
__global__ void cvt_w(WPtrs p, bf16* __restrict__ wiFM, bf16* __restrict__ whFM) {
  int idx = blockIdx.x * blockDim.x + threadIdx.x;     // 8 * 512 * 512 total
  int mat = idx >> 18;
  int rem = idx & 262143;
  int k = rem >> 9;
  int n = rem & 511;
  float v = p.w[mat][(size_t)k * K_ + n];
  int gate = mat & 3;
  size_t ncol_tile = (size_t)gate * 32 + (n >> 4);
  size_t frag = ncol_tile * 16 + (k >> 5);
  int lane = (n & 15) | (((k >> 3) & 1) << 4);
  int j = (k & 7) | (((k >> 4) & 1) << 3);
  bf16* dst = (mat < 4) ? wiFM : whFM;
  dst[(frag * 32 + lane) * 16 + j] = (bf16)v;
}

__global__ void cvt_b(BPtrs p, float* __restrict__ bias) {
  int idx = blockIdx.x * blockDim.x + threadIdx.x;     // 2048
  int g = idx >> 9, n = idx & 511;
  bias[idx] = p.bi[g][n] + p.bh[g][n];
}

// ---- Phase 1: xproj[t][b][gate*512+n] = x @ W_i + (b_i + b_h) -------------
// Grid: (NCOL/64, BT/128), 256 threads (8 waves). Wave = 32x32 tile (2x2 WMMA).

__global__ void __launch_bounds__(256)
gemm_xproj(const bf16* __restrict__ xb, const bf16* __restrict__ wFM,
           const float* __restrict__ bias, float* __restrict__ xproj) {
  int lane = threadIdx.x & 31;
  int wave = threadIdx.x >> 5;
  int blockN = blockIdx.x * 64;
  int blockM = blockIdx.y * 128;
  int m0 = blockM + (wave & 3) * 32;
  int n0 = blockN + (wave >> 2) * 32;

  v8f acc[2][2] = {};
  for (int k0 = 0; k0 < K_; k0 += 32) {
    v16bf a0 = load_a_frag(xb, K_, m0,      k0, lane);
    v16bf a1 = load_a_frag(xb, K_, m0 + 16, k0, lane);
    v16bf b0 = load_b_frag_fm(wFM, n0,      k0, lane);
    v16bf b1 = load_b_frag_fm(wFM, n0 + 16, k0, lane);
    acc[0][0] = wmma_bf16(a0, b0, acc[0][0]);
    acc[0][1] = wmma_bf16(a0, b1, acc[0][1]);
    acc[1][0] = wmma_bf16(a1, b0, acc[1][0]);
    acc[1][1] = wmma_bf16(a1, b1, acc[1][1]);
  }

  int bb = m0 >> 10;  // batch index; 32-row wave tile never crosses a T_ boundary
#pragma unroll
  for (int i = 0; i < 2; ++i) {
    int rb = m0 + i * 16 + ((lane >> 4) << 3);
#pragma unroll
    for (int j = 0; j < 2; ++j) {
      int col = n0 + j * 16 + (lane & 15);
      float bv = bias[col];
#pragma unroll
      for (int v = 0; v < 8; ++v) {
        int r  = rb + v;
        int tt = r & (T_ - 1);
        xproj[((size_t)tt * B_ + bb) * NCOL + col] = acc[i][j][v] + bv;
      }
    }
  }
}

// ---- Phase 2: persistent recurrent kernel --------------------------------
// 16 WGs x 256 threads. WG wg owns hidden cols [wg*32, wg*32+32).
// W_h slice (128 KB, fragment-major) is TDM-loaded into LDS once.
// Each step's xproj tile (32 KB) is TDM double-buffered through LDS,
// with the t+1 fetch overlapping step t's WMMA work + global barrier.

__global__ void __launch_bounds__(256, 1)
lstm_recur(const float* __restrict__ xproj, const bf16* __restrict__ whFM,
           bf16* __restrict__ hbuf, float* __restrict__ out,
           unsigned int* __restrict__ syncc) {
  extern __shared__ char smem[];
  const int tid  = threadIdx.x;
  const int lane = tid & 31;
  const int wave = tid >> 5;
  const int wg   = blockIdx.x;
  const int n_base = wg * HSLICE;
  const unsigned lds_base = (unsigned)(size_t)smem;   // LDS aperture low bits

  // ---- TDM preloads (wave 0 issues; TENSORcnt is per-wave) ----
  if (wave == 0) {
    // W_h slice: fragment-major global layout means each gate's chunk is a
    // contiguous 32 KB; 4 gates at 512 KB stride -> 2D tile of 8B elements.
    tdm_load(lds_base, (const char*)whFM + (size_t)wg * 32768,
             /*ds=8B*/3, /*td0*/4096, /*td1*/4, /*td2*/0,
             /*tile*/4096, 4, 0, /*s0*/65536ull, /*s1*/0ull);
    // xproj tile for t=0: 3D (32 cols x 64 batch x 4 gates) of f32,
    // row stride 2048, gate stride 512.
    tdm_load(lds_base + LDS_XBUF_OFF, xproj + wg * 32,
             /*ds=4B*/2, 32, 64, 4, 32, 64, 4, 2048ull, 512ull);
    __builtin_amdgcn_s_wait_tensorcnt(0);
  }
  __syncthreads();

  const int mt = wave & 3;
  const int nt = wave >> 2;
  const int colh  = n_base + nt * 16 + (lane & 15);  // hidden column
  const int rbase = mt * 16 + ((lane >> 4) << 3);    // batch row base (v adds 0..7)
  const int xcol  = nt * 16 + (lane & 15);           // col within 32-wide LDS tile

  v8f cstate = {};

  for (int t = 0; t < T_; ++t) {
    const bf16* hprev = hbuf + (size_t)(t & 1) * (B_ * K_);
    bf16*       hnext = hbuf + (size_t)((t + 1) & 1) * (B_ * K_);

    // init accumulators from LDS-staged input projections [gate][64][32] f32
    const float* xl =
        (const float*)(smem + LDS_XBUF_OFF + (size_t)(t & 1) * LDS_XBUF_BYTES);
    v8f acc[4];
#pragma unroll
    for (int g = 0; g < 4; ++g) {
#pragma unroll
      for (int v = 0; v < 8; ++v)
        acc[g][v] = xl[(size_t)g * 2048 + (rbase + v) * 32 + xcol];
    }

    // async-prefetch next step's xproj tile into the other LDS buffer;
    // the HBM fetch overlaps this step's WMMA work and the step barrier.
    if (wave == 0 && t + 1 < T_) {
      tdm_load(lds_base + LDS_XBUF_OFF + (unsigned)(((t + 1) & 1) * LDS_XBUF_BYTES),
               xproj + (size_t)(t + 1) * B_ * NCOL + wg * 32,
               2, 32, 64, 4, 32, 64, 4, 2048ull, 512ull);
    }

    // h_prev @ W_h  (K = 512 in 16 steps of 32); B frags from LDS (ds_load_b128)
#pragma unroll 4
    for (int ks = 0; ks < 16; ++ks) {
      v16bf a = load_a_frag(hprev, K_, mt * 16, ks * 32, lane);
#pragma unroll
      for (int g = 0; g < 4; ++g) {
        const v16bf* bp =
            (const v16bf*)(smem + (size_t)(((g * 2 + nt) * 16 + ks)) * 1024 +
                           lane * 32);
        acc[g] = wmma_bf16(a, *bp, acc[g]);
      }
    }

    // gates + state update (pure per-lane VALU; layouts already match)
#pragma unroll
    for (int v = 0; v < 8; ++v) {
      float iv = sigmoidf_(acc[0][v]);
      float fv = sigmoidf_(acc[1][v]);
      float gv = tanhf(acc[2][v]);
      float ov = sigmoidf_(acc[3][v]);
      float cv = fv * cstate[v] + iv * gv;
      cstate[v] = cv;
      float hv = ov * tanhf(cv);
      int b = rbase + v;
      hnext[(size_t)b * K_ + colh] = (bf16)hv;
      if (t == T_ - 1) {
        out[(size_t)b * K_ + colh] = hv;                       // h_t
        out[(size_t)(B_ * K_) + (size_t)b * K_ + colh] = cv;   // c_t
      }
    }

    // ---- device-wide step barrier ----
    __threadfence();        // make hnext visible at DEV scope
    __syncthreads();
    // Cluster barrier: NOP unless dispatched as a 16-WG cluster (safe either way).
    __builtin_amdgcn_s_cluster_barrier();
    if (tid == 0) {
      atomicAdd(syncc, 1u);
      unsigned target = (unsigned)(NWG_REC * (t + 1));
      volatile unsigned int* vs = syncc;
      while (*vs < target) { __builtin_amdgcn_s_sleep(1); }
    }
    __syncthreads();
    // complete the t+1 TDM prefetch before anyone reads the LDS buffer
    if (wave == 0) __builtin_amdgcn_s_wait_tensorcnt(0);
    __syncthreads();
    __threadfence();        // acquire other WGs' h slices
  }
}

// ---- host launch ----------------------------------------------------------

extern "C" void kernel_launch(void* const* d_in, const int* in_sizes, int n_in,
                              void* d_out, int out_size, void* d_ws, size_t ws_size,
                              hipStream_t stream) {
  const float* x = (const float*)d_in[0];
  WPtrs wp; BPtrs bp;
  for (int g = 0; g < 4; ++g) {
    wp.w[g]     = (const float*)d_in[1 + 4 * g];   // W_i{gate}
    wp.w[4 + g] = (const float*)d_in[2 + 4 * g];   // W_h{gate}
    bp.bi[g]    = (const float*)d_in[3 + 4 * g];
    bp.bh[g]    = (const float*)d_in[4 + 4 * g];
  }

  // workspace carve-up (all chunks naturally aligned)
  char* ws = (char*)d_ws;
  size_t off = 0;
  bf16*  xb    = (bf16*)(ws + off);  off += (size_t)BT_ * K_ * sizeof(bf16);       // 64 MB
  bf16*  wiFM  = (bf16*)(ws + off);  off += (size_t)NG * K_ * K_ * sizeof(bf16);   // 2 MB
  bf16*  whFM  = (bf16*)(ws + off);  off += (size_t)NG * K_ * K_ * sizeof(bf16);   // 2 MB
  float* bias  = (float*)(ws + off); off += (size_t)NCOL * sizeof(float);
  float* xproj = (float*)(ws + off); off += (size_t)T_ * B_ * NCOL * sizeof(float); // 512 MB
  bf16*  hbuf  = (bf16*)(ws + off);  off += (size_t)2 * B_ * K_ * sizeof(bf16);
  unsigned int* syncc = (unsigned int*)(ws + off);

  float* out = (float*)d_out;

  // 1) zero h ping-pong buffers + sync counter (ws is poisoned by harness)
  init_state<<<(2 * B_ * K_ * 2 / 4 + 255) / 256, 256, 0, stream>>>(
      (unsigned int*)hbuf, syncc);

  // 2) precision conversions / fragment-major weight swizzles
  cvt_x<<<(BT_ * K_ / 4) / 256, 256, 0, stream>>>(x, xb);
  cvt_w<<<(8 * K_ * K_) / 256, 256, 0, stream>>>(wp, wiFM, whFM);
  cvt_b<<<NCOL / 256, 256, 0, stream>>>(bp, bias);

  // 3) big input-projection GEMM (WMMA bf16, coalesced fragment-major B)
  dim3 ggrid(NCOL / 64, BT_ / 128);
  gemm_xproj<<<ggrid, 256, 0, stream>>>(xb, wiFM, bias, xproj);

  // 4) persistent recurrent phase: 16 WGs, 192 KB LDS each
  lstm_recur<<<NWG_REC, 256, LDS_TOTAL, stream>>>(xproj, whFM, hbuf, out, syncc);
}
